// MultiHeadAttention_79190607004265
// MI455X (gfx1250) — matrix operand order repaired and checked by reading, not compiled, not run on hardware
//
#include <hip/hip_runtime.h>
#include <hip/hip_bf16.h>

// MHA: B=4, S=2048, D_MODEL=1024, H=16, Dh=64, fp32 in/out.
// bf16 hi/lo pairs (3-WMMA fp32 emulation) on v_wmma_f32_16x16x32_bf16.
// All LDS staging via GLOBAL_LOAD_ASYNC_TO_LDS_B128 with double-buffered tiles.

#define DM     1024
#define NH     16
#define DH     64
#define BB     4
#define SS     2048
#define MTOT   (BB*SS)   // 8192

typedef unsigned short u16;
typedef __attribute__((ext_vector_type(16))) __bf16         v16bf;
typedef __attribute__((ext_vector_type(8)))  float          v8f;
typedef __attribute__((ext_vector_type(4)))  float          v4f;
typedef __attribute__((ext_vector_type(4)))  unsigned int   v4u;
typedef __attribute__((ext_vector_type(4)))  int            v4i;
typedef __attribute__((ext_vector_type(4)))  unsigned short v4us;

union FragBF {
  v16bf v;
  v4u   q[2];
  u16   us[16];
};

__device__ __forceinline__ u16 f32_bf16_rne(float f) {
  union { float f; unsigned u; } x; x.f = f;
  unsigned r = x.u + 0x7FFFu + ((x.u >> 16) & 1u);
  return (u16)(r >> 16);
}
__device__ __forceinline__ float bf16_f32(u16 h) {
  union { unsigned u; float f; } x; x.u = ((unsigned)h) << 16;
  return x.f;
}
__device__ __forceinline__ void split_bf16(float f, u16& hi, u16& lo) {
  hi = f32_bf16_rne(f);
  lo = f32_bf16_rne(f - bf16_f32(hi));
}

__device__ __forceinline__ v8f wmma_bf16(v16bf a, v16bf b, v8f c) {
  return __builtin_amdgcn_wmma_f32_16x16x32_bf16(false, a, false, b, (short)0, c, false, false);
}

// ---- async global->LDS staging (CDNA5), with compile-safe fallback --------
#if __has_builtin(__builtin_amdgcn_global_load_async_to_lds_b128)
#define USE_ASYNC 1
#else
#define USE_ASYNC 0
#endif

__device__ __forceinline__ void cp16(u16* lds, const u16* g) {
#if USE_ASYNC
  __builtin_amdgcn_global_load_async_to_lds_b128(
      (__attribute__((address_space(1))) v4i*)const_cast<u16*>(g),
      (__attribute__((address_space(3))) v4i*)lds, 0, 0);
#else
  *reinterpret_cast<v4u*>(lds) = *reinterpret_cast<const v4u*>(g);
#endif
}
__device__ __forceinline__ void async_fence() {
#if USE_ASYNC
#if __has_builtin(__builtin_amdgcn_s_wait_asynccnt)
  __builtin_amdgcn_s_wait_asynccnt(0);
#else
  asm volatile("s_wait_asynccnt 0x0" ::: "memory");
#endif
#endif
}

// A-matrix fragment (16x32 bf16, MxK), row = lane&15:
// lanes 0-15: elems 0-7 = K 0-7,  8-15 = K 16-23; lanes 16-31: K 8-15 / K 24-31.
__device__ __forceinline__ v16bf ld_a_frag(const u16* tile, int ld, int row,
                                           int kbase, int lane) {
  FragBF f;
  const u16* p = tile + row * ld + kbase + ((lane >> 4) << 3);
  f.q[0] = *reinterpret_cast<const v4u*>(p);
  f.q[1] = *reinterpret_cast<const v4u*>(p + 16);
  return f.v;
}
// B-matrix fragment (32x16 bf16, KxN) from an [N][K] LDS tile:
// lane col = lane&15; lanes 0-15: K 0-7 / K 16-23, lanes 16-31: K 8-15 / K 24-31.
__device__ __forceinline__ v16bf ld_b_frag(const u16* tile, int ld, int row,
                                           int kbase, int lane) {
  FragBF f;
  const u16* p = tile + row * ld + kbase + ((lane >> 4) << 3);
  f.q[0] = *reinterpret_cast<const v4u*>(p);
  f.q[1] = *reinterpret_cast<const v4u*>(p + 16);
  return f.v;
}

// ---------------------------------------------------------------------------
// fp32 -> bf16 hi/lo split converter (vectorized, n4 = element_count/4).
// ---------------------------------------------------------------------------
__global__ __launch_bounds__(256) void convert_split_kernel(const float* __restrict__ src,
                                                            u16* __restrict__ hi,
                                                            u16* __restrict__ lo,
                                                            int n4) {
  const int i = blockIdx.x * 256 + threadIdx.x;
  if (i >= n4) return;
  const v4f f = reinterpret_cast<const v4f*>(src)[i];
  v4us h, l;
#pragma unroll
  for (int j = 0; j < 4; j++) {
    u16 hh, ll;
    split_bf16(f[j], hh, ll);
    h[j] = hh; l[j] = ll;
  }
  reinterpret_cast<v4us*>(hi)[i] = h;
  reinterpret_cast<v4us*>(lo)[i] = l;
}

// ---------------------------------------------------------------------------
// C[m,n] = sum_k A[m,k]*W[n,k] on bf16 hi/lo pairs. M=8192, N=1024, K=1024.
// MODE 0: OF[m*1024+n] (fp32)
// MODE 1: OHi/OLo in [B,H,S,Dh]            (Q, K for attention)
// MODE 2: OHi/OLo in [B,H,Dh,S] transposed (V for attention B-fragments)
// Block 256 thr = 8 waves, 128x128 tile, wave 64x32, K-tile 64,
// double-buffered async LDS staging.
// ---------------------------------------------------------------------------
template <int MODE>
__global__ __launch_bounds__(256) void gemm_bf16pair_kernel(
    const u16* __restrict__ Ahi, const u16* __restrict__ Alo,
    const u16* __restrict__ Bhi, const u16* __restrict__ Blo,
    u16* __restrict__ OHi, u16* __restrict__ OLo, float* __restrict__ OF) {
  __shared__ alignas(16) u16 LAh[2][128 * 64];
  __shared__ alignas(16) u16 LAl[2][128 * 64];
  __shared__ alignas(16) u16 LBh[2][128 * 64];
  __shared__ alignas(16) u16 LBl[2][128 * 64];

  const int tid  = threadIdx.x;
  const int lane = tid & 31;
  const int wave = tid >> 5;
  const int mBlk = blockIdx.y * 128;
  const int nBlk = blockIdx.x * 128;
  const int wm   = (wave >> 2) * 64;
  const int wn   = (wave & 3) * 32;

  v8f acc[4][2];
#pragma unroll
  for (int i = 0; i < 4; i++)
#pragma unroll
    for (int j = 0; j < 2; j++)
#pragma unroll
      for (int r = 0; r < 8; r++) acc[i][j][r] = 0.0f;

  const int srow = tid >> 1;        // 0..127
  const int scol = (tid & 1) * 32;  // ushort offset (64B halves of a 128B row)

  auto stage = [&](int buf, int kt) {
    const size_t ga = (size_t)(mBlk + srow) * DM + kt + scol;
    const size_t gb = (size_t)(nBlk + srow) * DM + kt + scol;
    u16* lah = LAh[buf] + srow * 64 + scol;
    u16* lal = LAl[buf] + srow * 64 + scol;
    u16* lbh = LBh[buf] + srow * 64 + scol;
    u16* lbl = LBl[buf] + srow * 64 + scol;
#pragma unroll
    for (int c = 0; c < 32; c += 8) {
      cp16(lah + c, Ahi + ga + c);
      cp16(lal + c, Alo + ga + c);
      cp16(lbh + c, Bhi + gb + c);
      cp16(lbl + c, Blo + gb + c);
    }
  };

  stage(0, 0);
  async_fence();
  __syncthreads();

  int cur = 0;
  for (int kt = 0; kt < DM; kt += 64) {
    if (kt + 64 < DM) stage(cur ^ 1, kt + 64);  // DMA next tile during compute

#pragma unroll
    for (int ks = 0; ks < 64; ks += 32) {
      v16bf fbh[2], fbl[2];
#pragma unroll
      for (int ni = 0; ni < 2; ni++) {
        const int brow = wn + ni * 16 + (lane & 15);
        fbh[ni] = ld_b_frag(LBh[cur], 64, brow, ks, lane);
        fbl[ni] = ld_b_frag(LBl[cur], 64, brow, ks, lane);
      }
#pragma unroll
      for (int mi = 0; mi < 4; mi++) {
        const int arow = wm + mi * 16 + (lane & 15);
        v16bf fah = ld_a_frag(LAh[cur], 64, arow, ks, lane);
        v16bf fal = ld_a_frag(LAl[cur], 64, arow, ks, lane);
#pragma unroll
        for (int ni = 0; ni < 2; ni++) {
          acc[mi][ni] = wmma_bf16(fah, fbh[ni], acc[mi][ni]);
          acc[mi][ni] = wmma_bf16(fah, fbl[ni], acc[mi][ni]);
          acc[mi][ni] = wmma_bf16(fal, fbh[ni], acc[mi][ni]);
        }
      }
    }
    async_fence();
    __syncthreads();
    cur ^= 1;
  }

  // C layout: lane L -> n = L&15, m = r + 8*(L>=16)
#pragma unroll
  for (int mi = 0; mi < 4; mi++) {
#pragma unroll
    for (int ni = 0; ni < 2; ni++) {
      const int col   = nBlk + wn + ni * 16 + (lane & 15);
      const int rbase = mBlk + wm + mi * 16 + ((lane >> 4) << 3);
#pragma unroll
      for (int r = 0; r < 8; r++) {
        const int m = rbase + r;
        const float val = acc[mi][ni][r];
        if (MODE == 0) {
          OF[(size_t)m * DM + col] = val;
        } else {
          const int b = m >> 11, s = m & 2047;
          const int hd = col >> 6, dh = col & 63;
          u16 h, l;
          split_bf16(val, h, l);
          size_t idx;
          if (MODE == 1) idx = ((size_t)((b << 4) | hd) * SS + s) * DH + dh;
          else           idx = ((size_t)((b << 4) | hd) * DH + dh) * SS + s;
          OHi[idx] = h; OLo[idx] = l;
        }
      }
    }
  }
}

// ---------------------------------------------------------------------------
// Flash attention: block = one (b,h) x one 64-row Q block; KV chunks of 64,
// double-buffered async K/V staging.
// Q/K: [BH][S][64] hi/lo;  V: [BH][64][S] hi/lo (pre-transposed).
// 8 waves: wave w owns tiles rows (w>>1)*16, cols (w&1)*32.
// ---------------------------------------------------------------------------
__global__ __launch_bounds__(256) void flash_attn_kernel(
    const u16* __restrict__ Qhi, const u16* __restrict__ Qlo,
    const u16* __restrict__ Khi, const u16* __restrict__ Klo,
    const u16* __restrict__ Vthi, const u16* __restrict__ Vtlo,
    u16* __restrict__ Chi, u16* __restrict__ Clo) {
  __shared__ alignas(16) u16 Qh[64 * 64];
  __shared__ alignas(16) u16 Ql[64 * 64];
  __shared__ alignas(16) u16 Kh[2][64 * 64];
  __shared__ alignas(16) u16 Kl[2][64 * 64];
  __shared__ alignas(16) u16 Vh[2][64 * 64];  // [dh][kv]
  __shared__ alignas(16) u16 Vl[2][64 * 64];
  __shared__ alignas(16) u16 Ph[64 * 64];
  __shared__ alignas(16) u16 Pl[64 * 64];
  __shared__ alignas(16) float Sarr[64 * 64];
  __shared__ float m_s[64], l_s[64], f_s[64];

  const int tid  = threadIdx.x;
  const int lane = tid & 31;
  const int wave = tid >> 5;
  const int bh   = blockIdx.y;
  const int q0   = blockIdx.x * 64;

  const int tm = (wave >> 1) * 16;
  const int tn = (wave & 1) * 32;

  const int row = tid >> 2;        // 0..63
  const int cof = (tid & 3) * 16;  // 0..48 ushorts (32B chunks)

  auto stage_kv = [&](int buf, int kv0) {
    const size_t gk = ((size_t)bh * SS + kv0 + row) * DH + cof;
    const size_t gv = ((size_t)bh * DH + row) * SS + kv0 + cof;
    cp16(Kh[buf] + row * 64 + cof,     Khi + gk);
    cp16(Kh[buf] + row * 64 + cof + 8, Khi + gk + 8);
    cp16(Kl[buf] + row * 64 + cof,     Klo + gk);
    cp16(Kl[buf] + row * 64 + cof + 8, Klo + gk + 8);
    cp16(Vh[buf] + row * 64 + cof,     Vthi + gv);
    cp16(Vh[buf] + row * 64 + cof + 8, Vthi + gv + 8);
    cp16(Vl[buf] + row * 64 + cof,     Vtlo + gv);
    cp16(Vl[buf] + row * 64 + cof + 8, Vtlo + gv + 8);
  };

  {  // stage Q block + first KV chunk
    const size_t gq = ((size_t)bh * SS + q0 + row) * DH + cof;
    cp16(Qh + row * 64 + cof,     Qhi + gq);
    cp16(Qh + row * 64 + cof + 8, Qhi + gq + 8);
    cp16(Ql + row * 64 + cof,     Qlo + gq);
    cp16(Ql + row * 64 + cof + 8, Qlo + gq + 8);
    stage_kv(0, 0);
  }
  if (tid < 64) { m_s[tid] = -1e30f; l_s[tid] = 0.0f; }

  v8f ctx[2];
#pragma unroll
  for (int j = 0; j < 2; j++)
#pragma unroll
    for (int r = 0; r < 8; r++) ctx[j][r] = 0.0f;

  async_fence();
  __syncthreads();

  int cur = 0;
  for (int kv0 = 0; kv0 < SS; kv0 += 64) {
    if (kv0 + 64 < SS) stage_kv(cur ^ 1, kv0 + 64);  // DMA next chunk now

    // S = (Q K^T) / sqrt(64)
    v8f s[2];
#pragma unroll
    for (int j = 0; j < 2; j++)
#pragma unroll
      for (int r = 0; r < 8; r++) s[j][r] = 0.0f;

#pragma unroll
    for (int ks = 0; ks < 64; ks += 32) {
      v16bf fah = ld_a_frag(Qh, 64, tm + (lane & 15), ks, lane);
      v16bf fal = ld_a_frag(Ql, 64, tm + (lane & 15), ks, lane);
#pragma unroll
      for (int ni = 0; ni < 2; ni++) {
        const int brow = tn + ni * 16 + (lane & 15);
        v16bf fbh = ld_b_frag(Kh[cur], 64, brow, ks, lane);
        v16bf fbl = ld_b_frag(Kl[cur], 64, brow, ks, lane);
        s[ni] = wmma_bf16(fah, fbh, s[ni]);
        s[ni] = wmma_bf16(fah, fbl, s[ni]);
        s[ni] = wmma_bf16(fal, fbh, s[ni]);
      }
    }
#pragma unroll
    for (int ni = 0; ni < 2; ni++)
#pragma unroll
      for (int r = 0; r < 8; r++)
        Sarr[(tm + r + ((lane >> 4) << 3)) * 64 + tn + ni * 16 + (lane & 15)] =
            s[ni][r] * 0.125f;
    __syncthreads();

    // online softmax: thread t < 64 owns row t
    if (tid < 64) {
      const float* srow = Sarr + tid * 64;
      float mx = m_s[tid];
#pragma unroll 8
      for (int j = 0; j < 64; j++) mx = fmaxf(mx, srow[j]);
      const float corr = __expf(m_s[tid] - mx);
      float l = l_s[tid] * corr;
#pragma unroll 8
      for (int j = 0; j < 64; j++) {
        const float p = __expf(srow[j] - mx);
        l += p;
        u16 h, lo;
        split_bf16(p, h, lo);
        Ph[tid * 64 + j] = h;
        Pl[tid * 64 + j] = lo;
      }
      m_s[tid] = mx; l_s[tid] = l; f_s[tid] = corr;
    }
    __syncthreads();

    // ctx = ctx*corr + P @ V
#pragma unroll
    for (int r = 0; r < 8; r++) {
      const float sc = f_s[tm + r + ((lane >> 4) << 3)];
      ctx[0][r] *= sc;
      ctx[1][r] *= sc;
    }
#pragma unroll
    for (int ks = 0; ks < 64; ks += 32) {
      v16bf fah = ld_a_frag(Ph, 64, tm + (lane & 15), ks, lane);
      v16bf fal = ld_a_frag(Pl, 64, tm + (lane & 15), ks, lane);
#pragma unroll
      for (int ni = 0; ni < 2; ni++) {
        const int brow = tn + ni * 16 + (lane & 15);
        v16bf fbh = ld_b_frag(Vh[cur], 64, brow, ks, lane);
        v16bf fbl = ld_b_frag(Vl[cur], 64, brow, ks, lane);
        ctx[ni] = wmma_bf16(fah, fbh, ctx[ni]);
        ctx[ni] = wmma_bf16(fah, fbl, ctx[ni]);
        ctx[ni] = wmma_bf16(fal, fbh, ctx[ni]);
      }
    }
    async_fence();
    __syncthreads();
    cur ^= 1;
  }

  // normalize, split, store ctx into [B,S,D_MODEL] hi/lo for the O projection
  const int b = bh >> 4, hd = bh & 15;
#pragma unroll
  for (int ni = 0; ni < 2; ni++) {
#pragma unroll
    for (int r = 0; r < 8; r++) {
      const int qrow = tm + r + ((lane >> 4) << 3);
      const int dh   = tn + ni * 16 + (lane & 15);
      const float inv = 1.0f / l_s[qrow];
      u16 h, l;
      split_bf16(ctx[ni][r] * inv, h, l);
      const size_t idx = ((size_t)b * SS + q0 + qrow) * DM + hd * DH + dh;
      Chi[idx] = h; Clo[idx] = l;
    }
  }
}

// ---------------------------------------------------------------------------
extern "C" void kernel_launch(void* const* d_in, const int* in_sizes, int n_in,
                              void* d_out, int out_size, void* d_ws, size_t ws_size,
                              hipStream_t stream) {
  const float* q_in  = (const float*)d_in[0];
  const float* kv_in = (const float*)d_in[1];
  const float* Wq = (const float*)d_in[2];
  const float* Wk = (const float*)d_in[3];
  const float* Wv = (const float*)d_in[4];
  const float* Wo = (const float*)d_in[5];
  float* out = (float*)d_out;

  char* ws = (char*)d_ws;
  auto alloc = [&](size_t bytes) -> u16* {
    u16* p = (u16*)ws;
    ws += (bytes + 255) & ~(size_t)255;
    return p;
  };
  const size_t wbytes = (size_t)DM * DM * sizeof(u16);    // 2 MB
  const size_t abytes = (size_t)MTOT * DM * sizeof(u16);  // 16 MB

  u16 *WqH = alloc(wbytes), *WqL = alloc(wbytes);
  u16 *WkH = alloc(wbytes), *WkL = alloc(wbytes);
  u16 *WvH = alloc(wbytes), *WvL = alloc(wbytes);
  u16 *WoH = alloc(wbytes), *WoL = alloc(wbytes);
  u16 *XqH = alloc(abytes), *XqL = alloc(abytes);
  u16 *XkH = alloc(abytes), *XkL = alloc(abytes);
  u16 *QH  = alloc(abytes), *QL  = alloc(abytes);
  u16 *KH  = alloc(abytes), *KL  = alloc(abytes);
  u16 *VH  = alloc(abytes), *VL  = alloc(abytes);
  u16 *CH  = alloc(abytes), *CL  = alloc(abytes);

  dim3 blk(256);
  const int wn4 = DM * DM / 4, an4 = MTOT * DM / 4;
  convert_split_kernel<<<wn4 / 256, blk, 0, stream>>>(Wq, WqH, WqL, wn4);
  convert_split_kernel<<<wn4 / 256, blk, 0, stream>>>(Wk, WkH, WkL, wn4);
  convert_split_kernel<<<wn4 / 256, blk, 0, stream>>>(Wv, WvH, WvL, wn4);
  convert_split_kernel<<<wn4 / 256, blk, 0, stream>>>(Wo, WoH, WoL, wn4);
  convert_split_kernel<<<an4 / 256, blk, 0, stream>>>(q_in,  XqH, XqL, an4);
  convert_split_kernel<<<an4 / 256, blk, 0, stream>>>(kv_in, XkH, XkL, an4);

  dim3 gblk(DM / 128, MTOT / 128);  // (8, 64)
  gemm_bf16pair_kernel<1><<<gblk, blk, 0, stream>>>(XqH, XqL, WqH, WqL, QH, QL, nullptr);
  gemm_bf16pair_kernel<1><<<gblk, blk, 0, stream>>>(XkH, XkL, WkH, WkL, KH, KL, nullptr);
  gemm_bf16pair_kernel<2><<<gblk, blk, 0, stream>>>(XkH, XkL, WvH, WvL, VH, VL, nullptr);

  dim3 gatt(SS / 64, BB * NH);      // (32, 64)
  flash_attn_kernel<<<gatt, blk, 0, stream>>>(QH, QL, KH, KL, VH, VL, CH, CL);

  gemm_bf16pair_kernel<0><<<gblk, blk, 0, stream>>>(CH, CL, WoH, WoL, nullptr, nullptr, out);
}
